// GNMA_20126216749376
// MI455X (gfx1250) — matrix operand order repaired and checked by reading, not compiled, not run on hardware
//
#include <hip/hip_runtime.h>
#include <hip/hip_bf16.h>
#include <math.h>

// ---------------------------------------------------------------------------
// GNM attention pipeline for MI455X (gfx1250, wave32, WMMA).
//   1) k_emb      : RBF(d_ij) -> MLP(64->64 relu ->1) via v_wmma_f32_16x16x32_f16
//   2) k_deg/k_lap: degree + Laplacian (in place), row-abs-sum -> ||A||_inf
//   3) k_scale    : X0 = lap / ||A||_1*||A||_inf   (lap symmetric => lap^T = lap)
//   4) k_gemm_ns  : Newton-Schulz pinv iterations, fp32 WMMA (16x16x4_f32),
//                   128x64 block tile, KC=32 double-buffered LDS staging via
//                   GLOBAL_LOAD_ASYNC_TO_LDS_B128 (ASYNCcnt), 8 waves/block,
//                   each wave a 16x64 strip (A-frag reuse x4).
//   5) k_attn     : row softmax of corr, weighted coord aggregation -> (N,3)
// Workspace: 4*N*N f32 (64 MB) + deg + 1 scalar.
// ---------------------------------------------------------------------------

#define NATOMS 2048
#define DIM 64
#define CUTOFF 10.0f
#define NS_ITERS 32

#define MB 128
#define NBLK 64
#define KC 32
#define KPAD 4   // row stride 36 floats = 144 B: keeps 16B alignment for B128
                 // async stores and spreads rows across LDS banks

typedef __attribute__((ext_vector_type(16))) _Float16 v16h;
typedef __attribute__((ext_vector_type(8)))  float    v8f;
typedef __attribute__((ext_vector_type(2)))  float    v2f;

// ---------------------------------------------------------------------------
// Kernel 1: per-pair RBF expansion + 2-layer MLP -> emb[i][j].
// One wave handles 16 pairs (fixed i, 16 consecutive j). The (16 x 64) rbf
// matrix times W1 (64 x 64) is two v_wmma_f32_16x16x32_f16 per 16-col tile.
// Layer 2 (64 -> 1) is folded into the epilogue with a 16-lane xor reduction.
// ---------------------------------------------------------------------------
__global__ __launch_bounds__(256) void k_emb(
    const float* __restrict__ coords, const float* __restrict__ W1,
    const float* __restrict__ b1, const float* __restrict__ W2,
    const float* __restrict__ b2, float* __restrict__ emb)
{
  const int tid  = threadIdx.x;
  const int wave = tid >> 5;
  const int lane = tid & 31;
  const int m    = lane & 15;   // pair (row M) within wave
  const int hi   = lane >> 4;   // half-wave: selects K-slice per ISA layout

  const int i  = blockIdx.y;
  const int j0 = blockIdx.x * 128 + wave * 16;
  const int j  = j0 + m;

  const float dx = coords[3*i+0] - coords[3*j+0];
  const float dy = coords[3*i+1] - coords[3*j+1];
  const float dz = coords[3*i+2] - coords[3*j+2];
  const float d  = sqrtf(fmaxf(dx*dx + dy*dy + dz*dz, 1e-12f));

  const float w    = CUTOFF / (float)(DIM - 1);
  const float invw = (float)(DIM - 1) / CUTOFF;

  // A fragments (16x32 f16 each, K=0..31 and K=32..63).
  // ISA layout: lanes 0-15 hold K {0..7} U {16..23}, lanes 16-31 K {8..15} U {24..31}
  // with halves h: v = h/2, K = (v<4 ? 2v+p : 16+2(v-4)+p) + 8*hi.
  v16h aLo, aHi;
  #pragma unroll
  for (int h = 0; h < 16; ++h) {
    int v  = h >> 1, p = h & 1;
    int kl = (v < 4) ? (2*v + p) : (16 + 2*(v - 4) + p);
    int k  = kl + hi * 8;
    float t0 = (d - (float)k * w) * invw;
    aLo[h] = (_Float16)__expf(-t0 * t0);
    float t1 = (d - (float)(k + 32) * w) * invw;
    aHi[h] = (_Float16)__expf(-t1 * t1);
  }

  float acc[8];
  #pragma unroll
  for (int r = 0; r < 8; ++r) acc[r] = 0.f;

  #pragma unroll
  for (int nt = 0; nt < 4; ++nt) {           // 4 column tiles of W1
    // B fragments: lane holds column n = nt*16+m; K = (h<8 ? h : 8+h) + 8*hi per 32-K chunk.
    v16h bLo, bHi;
    #pragma unroll
    for (int h = 0; h < 16; ++h) {
      int k = (h < 8) ? (8 * hi + h) : (16 + 8 * hi + (h - 8));
      bLo[h] = (_Float16)W1[(size_t)k * DIM + nt * 16 + m];
      bHi[h] = (_Float16)W1[(size_t)(k + 32) * DIM + nt * 16 + m];
    }
    v8f c = {};
    c = __builtin_amdgcn_wmma_f32_16x16x32_f16(false, aLo, false, bLo,
                                               (short)0, c, false, false);
    c = __builtin_amdgcn_wmma_f32_16x16x32_f16(false, aHi, false, bHi,
                                               (short)0, c, false, false);
    // epilogue of layer 1 + layer 2 partial: lane owns column N = nt*16+m
    const float b1v = b1[nt * 16 + m];
    const float w2v = W2[nt * 16 + m];
    #pragma unroll
    for (int r = 0; r < 8; ++r) {
      float hval = fmaxf(c[r] + b1v, 0.f);   // relu
      acc[r] += hval * w2v;
    }
  }

  // reduce over the 16 columns held by this half-wave; C-tile row M = r + 8*hi
  const float b2v = b2[0];
  #pragma unroll
  for (int r = 0; r < 8; ++r) {
    float v = acc[r];
    #pragma unroll
    for (int msk = 8; msk; msk >>= 1) v += __shfl_xor(v, msk, 32);
    if (m == 0)
      emb[(size_t)i * NATOMS + j0 + r + 8 * hi] = v + b2v;
  }
}

// ---------------------------------------------------------------------------
// Kernel 2: row sums of emb -> deg
// ---------------------------------------------------------------------------
__global__ __launch_bounds__(256) void k_deg(const float* __restrict__ emb,
                                             float* __restrict__ deg)
{
  __shared__ float sh[8];
  const int i = blockIdx.x;
  float s = 0.f;
  for (int j = threadIdx.x; j < NATOMS; j += 256)
    s += emb[(size_t)i * NATOMS + j];
  #pragma unroll
  for (int msk = 16; msk; msk >>= 1) s += __shfl_xor(s, msk, 32);
  if ((threadIdx.x & 31) == 0) sh[threadIdx.x >> 5] = s;
  __syncthreads();
  if (threadIdx.x == 0) {
    float t = 0.f;
    #pragma unroll
    for (int q = 0; q < 8; ++q) t += sh[q];
    deg[i] = t;
  }
}

// ---------------------------------------------------------------------------
// Kernel 3: lap = diag(deg) - emb (in place), plus max abs-row-sum via
// atomic max on float bits (valid: abs sums are non-negative).
// ---------------------------------------------------------------------------
__global__ __launch_bounds__(256) void k_lap(float* __restrict__ a,
                                             const float* __restrict__ deg,
                                             unsigned* __restrict__ mxbits)
{
  __shared__ float sh[8];
  const int i = blockIdx.x;
  const float di = deg[i];
  float s = 0.f;
  for (int j = threadIdx.x; j < NATOMS; j += 256) {
    float e = a[(size_t)i * NATOMS + j];
    float l = (j == i ? di : 0.f) - e;
    a[(size_t)i * NATOMS + j] = l;
    s += fabsf(l);
  }
  #pragma unroll
  for (int msk = 16; msk; msk >>= 1) s += __shfl_xor(s, msk, 32);
  if ((threadIdx.x & 31) == 0) sh[threadIdx.x >> 5] = s;
  __syncthreads();
  if (threadIdx.x == 0) {
    float t = 0.f;
    #pragma unroll
    for (int q = 0; q < 8; ++q) t += sh[q];
    atomicMax(mxbits, __float_as_uint(t));
  }
}

// ---------------------------------------------------------------------------
// Kernel 4: X0 = lap * 1/(||A||_1 ||A||_inf)  (symmetric: ||A||_1 == ||A||_inf)
// ---------------------------------------------------------------------------
__global__ __launch_bounds__(256) void k_scale(const float* __restrict__ lap,
                                               const unsigned* __restrict__ mxbits,
                                               float* __restrict__ X)
{
  const float mx = __uint_as_float(*mxbits);
  const float alpha = 1.f / fmaxf(mx * mx, 1e-30f);
  const size_t idx = (size_t)blockIdx.x * 256 + threadIdx.x;
  X[idx] = alpha * lap[idx];
}

// ---------------------------------------------------------------------------
// Kernel 5: fp32 WMMA GEMM for Newton-Schulz.
//   mode 0: D = A @ B
//   mode 1: D = 2*Csrc - A @ B      (fused X_{k+1} = 2X - X(AX))
// Block tile 128x64, K chunk 32. Double-buffered LDS staging with
// GLOBAL_LOAD_ASYNC_TO_LDS_B128: chunk k+1 DMAs into the alternate buffer
// while the 32 WMMAs of chunk k execute; one s_wait_asynccnt + barrier per
// chunk. LDS address = low 32 bits of the generic shared pointer (ISA flat->
// LDS truncation rule); all chunks are 16B aligned (row stride 144 B / 256 B).
// ---------------------------------------------------------------------------
__global__ __launch_bounds__(256) void k_gemm_ns(
    const float* __restrict__ A, const float* __restrict__ B,
    const float* __restrict__ Csrc, float* __restrict__ D, int mode)
{
  __shared__ float As[2][MB][KC + KPAD];
  __shared__ float Bs[2][KC][NBLK];

  const int tid  = threadIdx.x;
  const int wave = tid >> 5;
  const int lane = tid & 31;
  const int m    = lane & 15;
  const int hi   = lane >> 4;
  const int bm   = blockIdx.y * MB;
  const int bn   = blockIdx.x * NBLK;

  v8f acc[4];
  #pragma unroll
  for (int nt = 0; nt < 4; ++nt) acc[nt] = (v8f){};

  // Issue the async global->LDS DMA for one K chunk into buffer `buf`.
  auto issue = [&](int kc, int buf) {
    // A tile (128 x 32 floats) = 1024 x B128 transfers; 4 per thread
    #pragma unroll
    for (int q = 0; q < 4; ++q) {
      int fidx = q * 256 + tid;
      int row  = fidx >> 3;
      int c4   = fidx & 7;
      const float* g = &A[(size_t)(bm + row) * NATOMS + kc + c4 * 4];
      unsigned l = (unsigned)(size_t)&As[buf][row][c4 * 4];
      asm volatile("global_load_async_to_lds_b128 %0, %1, off"
                   :: "v"(l), "v"(g) : "memory");
    }
    // B tile (32 x 64 floats) = 512 x B128 transfers; 2 per thread
    #pragma unroll
    for (int q = 0; q < 2; ++q) {
      int fidx = q * 256 + tid;
      int row  = fidx >> 4;
      int c4   = fidx & 15;
      const float* g = &B[(size_t)(kc + row) * NATOMS + bn + c4 * 4];
      unsigned l = (unsigned)(size_t)&Bs[buf][row][c4 * 4];
      asm volatile("global_load_async_to_lds_b128 %0, %1, off"
                   :: "v"(l), "v"(g) : "memory");
    }
  };

  issue(0, 0);
  asm volatile("s_wait_asynccnt 0x0" ::: "memory");
  __syncthreads();

  for (int kc = 0; kc < NATOMS; kc += KC) {
    const int buf = (kc / KC) & 1;
    if (kc + KC < NATOMS) issue(kc + KC, buf ^ 1);   // overlap DMA with WMMA

    #pragma unroll
    for (int kk = 0; kk < KC; kk += 4) {
      // A frag 16x4: lanes 0-15 K={0,1}, lanes 16-31 K={2,3}
      v2f a;
      a.x = As[buf][wave * 16 + m][kk + 2 * hi + 0];
      a.y = As[buf][wave * 16 + m][kk + 2 * hi + 1];
      #pragma unroll
      for (int nt = 0; nt < 4; ++nt) {
        v2f b;
        b.x = Bs[buf][kk + 2 * hi + 0][nt * 16 + m];
        b.y = Bs[buf][kk + 2 * hi + 1][nt * 16 + m];
        acc[nt] = __builtin_amdgcn_wmma_f32_16x16x4_f32(
            false, a, false, b, (short)0, acc[nt], false, false);
      }
    }

    asm volatile("s_wait_asynccnt 0x0" ::: "memory");
    __syncthreads();
  }

  // epilogue: C/D layout row M = r + 8*hi, col N = m
  const int rowbase = bm + wave * 16 + 8 * hi;
  #pragma unroll
  for (int nt = 0; nt < 4; ++nt) {
    const int col = bn + nt * 16 + m;
    #pragma unroll
    for (int r = 0; r < 8; ++r) {
      size_t off = (size_t)(rowbase + r) * NATOMS + col;
      float v = acc[nt][r];
      if (mode) v = 2.f * Csrc[off] - v;
      D[off] = v;
    }
  }
}

// ---------------------------------------------------------------------------
// Kernel 6: row softmax of corr + weighted aggregation of coords -> (N,3)
// ---------------------------------------------------------------------------
__global__ __launch_bounds__(256) void k_attn(const float* __restrict__ corr,
                                              const float* __restrict__ coords,
                                              float* __restrict__ out)
{
  __shared__ float sh[8];
  __shared__ float sh4[4][8];
  const int i = blockIdx.x;
  const int tid = threadIdx.x;

  float mx = -3.402823466e38f;
  for (int j = tid; j < NATOMS; j += 256)
    mx = fmaxf(mx, corr[(size_t)i * NATOMS + j]);
  #pragma unroll
  for (int msk = 16; msk; msk >>= 1) mx = fmaxf(mx, __shfl_xor(mx, msk, 32));
  if ((tid & 31) == 0) sh[tid >> 5] = mx;
  __syncthreads();
  mx = sh[0];
  #pragma unroll
  for (int q = 1; q < 8; ++q) mx = fmaxf(mx, sh[q]);
  __syncthreads();

  float s = 0.f, cx = 0.f, cy = 0.f, cz = 0.f;
  for (int j = tid; j < NATOMS; j += 256) {
    float e = __expf(corr[(size_t)i * NATOMS + j] - mx);
    s  += e;
    cx += e * coords[3 * j + 0];
    cy += e * coords[3 * j + 1];
    cz += e * coords[3 * j + 2];
  }
  #pragma unroll
  for (int msk = 16; msk; msk >>= 1) {
    s  += __shfl_xor(s,  msk, 32);
    cx += __shfl_xor(cx, msk, 32);
    cy += __shfl_xor(cy, msk, 32);
    cz += __shfl_xor(cz, msk, 32);
  }
  if ((tid & 31) == 0) {
    sh4[0][tid >> 5] = s;  sh4[1][tid >> 5] = cx;
    sh4[2][tid >> 5] = cy; sh4[3][tid >> 5] = cz;
  }
  __syncthreads();
  if (tid == 0) {
    float S = 0.f, CX = 0.f, CY = 0.f, CZ = 0.f;
    #pragma unroll
    for (int q = 0; q < 8; ++q) {
      S += sh4[0][q]; CX += sh4[1][q]; CY += sh4[2][q]; CZ += sh4[3][q];
    }
    out[3 * i + 0] = CX / S;
    out[3 * i + 1] = CY / S;
    out[3 * i + 2] = CZ / S;
  }
}

// ---------------------------------------------------------------------------
extern "C" void kernel_launch(void* const* d_in, const int* in_sizes, int n_in,
                              void* d_out, int out_size, void* d_ws, size_t ws_size,
                              hipStream_t stream)
{
  (void)in_sizes; (void)n_in; (void)out_size; (void)ws_size;
  const float* coords = (const float*)d_in[0];
  const float* W1     = (const float*)d_in[1];
  const float* b1     = (const float*)d_in[2];
  const float* W2     = (const float*)d_in[3];
  const float* b2     = (const float*)d_in[4];
  float* out = (float*)d_out;

  const size_t NN = (size_t)NATOMS * NATOMS;
  float* lap = (float*)d_ws;       // emb, then Laplacian in place
  float* X   = lap + NN;
  float* Y   = X + NN;
  float* X2  = Y + NN;
  float* deg = X2 + NN;
  unsigned* mx = (unsigned*)(deg + NATOMS);

  hipMemsetAsync(mx, 0, sizeof(unsigned), stream);

  k_emb<<<dim3(NATOMS / 128, NATOMS), 256, 0, stream>>>(coords, W1, b1, W2, b2, lap);
  k_deg<<<NATOMS, 256, 0, stream>>>(lap, deg);
  k_lap<<<NATOMS, 256, 0, stream>>>(lap, deg, mx);
  k_scale<<<(int)(NN / 256), 256, 0, stream>>>(lap, mx, X);

  dim3 gg(NATOMS / NBLK, NATOMS / MB);
  float* xa = X;
  float* xb = X2;
  for (int it = 0; it < NS_ITERS; ++it) {
    k_gemm_ns<<<gg, 256, 0, stream>>>(lap, xa, nullptr, Y, 0); // Y = A @ X
    k_gemm_ns<<<gg, 256, 0, stream>>>(xa, Y, xa, xb, 1);       // X' = 2X - X@Y
    float* t = xa; xa = xb; xb = t;
  }

  k_attn<<<NATOMS, 256, 0, stream>>>(xa, coords, out);
}
